// self_attn_36378372997813
// MI455X (gfx1250) — matrix operand order repaired and checked by reading, not compiled, not run on hardware
//
#include <hip/hip_runtime.h>
#include <hip/hip_bf16.h>

// ---------------------------------------------------------------------------
// Problem sizes (fixed by the reference)
// ---------------------------------------------------------------------------
#define BB      8
#define SS      2048
#define INDIM   1024
#define HID     1024
#define MTOT    (BB * SS)          // 16384 rows for GEMM1

typedef __attribute__((ext_vector_type(16))) __bf16 v16bf;
typedef __attribute__((ext_vector_type(8)))  __bf16 v8bf;
typedef __attribute__((ext_vector_type(4)))  __bf16 v4bf;
typedef __attribute__((ext_vector_type(8)))  float  v8f;

// round-to-nearest-even f32 -> bf16
__device__ __forceinline__ __bf16 f2bf(float x) {
    unsigned u = __builtin_bit_cast(unsigned, x);
    u += 0x7fffu + ((u >> 16) & 1u);
    return __builtin_bit_cast(__bf16, (unsigned short)(u >> 16));
}

// Build the 16x32 bf16 A fragment for one lane.
// Layout: lane L (row = L&15, hi = L>>4):
//   VGPR0..7 hold K = kb+0..kb+15, kb = hi?16:0 (contiguous, matches B).
__device__ __forceinline__ v16bf load_a_frag(const __bf16* p) {
    v8bf lo = *(const v8bf*)(p);        // K = kb .. kb+7
    v8bf hi = *(const v8bf*)(p + 8);    // K = kb+8 .. kb+15
    return __builtin_shufflevector(lo, hi, 0,1,2,3,4,5,6,7,8,9,10,11,12,13,14,15);
}

__device__ __forceinline__ v8f wmma_bf16(v16bf a, v16bf b, v8f c) {
    return __builtin_amdgcn_wmma_f32_16x16x32_bf16(
        false, a, false, b, (short)0, c, false, false);
}

// ---------------------------------------------------------------------------
// fp32 -> bf16 conversion (4 elements / thread)
// ---------------------------------------------------------------------------
__global__ void cvt_bf16_kernel(const float* __restrict__ src,
                                __bf16* __restrict__ dst, int n4) {
    int i = blockIdx.x * blockDim.x + threadIdx.x;
    if (i >= n4) return;
    const float4 f = ((const float4*)src)[i];
    v4bf o; o.x = f2bf(f.x); o.y = f2bf(f.y); o.z = f2bf(f.z); o.w = f2bf(f.w);
    ((v4bf*)dst)[i] = o;
}

// out[b*H + x] = lin_b  (atomic accumulators initialized here)
__global__ void init_out_kernel(float* __restrict__ out,
                                const float* __restrict__ lin_b) {
    int i = blockIdx.x * blockDim.x + threadIdx.x;
    if (i < BB * HID) out[i] = lin_b[0];
}

// ---------------------------------------------------------------------------
// Fragment group helpers for the 32(M) x 64(N) wave tile
// ---------------------------------------------------------------------------
template<int LD>
__device__ __forceinline__ void load_frags(const __bf16* __restrict__ aPtr0,
                                           const __bf16* __restrict__ aPtr1,
                                           const __bf16* __restrict__ bPtr,
                                           int k0, v16bf a[2], v16bf b[4]) {
    a[0] = load_a_frag(aPtr0 + k0);
    a[1] = load_a_frag(aPtr1 + k0);
    #pragma unroll
    for (int t = 0; t < 4; ++t)
        b[t] = *(const v16bf*)(bPtr + (size_t)t * 16 * LD + k0);
}

__device__ __forceinline__ void compute8(v8f acc[2][4],
                                         const v16bf a[2], const v16bf b[4]) {
    #pragma unroll
    for (int m = 0; m < 2; ++m)
        #pragma unroll
        for (int t = 0; t < 4; ++t)
            acc[m][t] = wmma_bf16(a[m], b[t], acc[m][t]);
}

// ---------------------------------------------------------------------------
// Ping-pong pipelined K loop (64 K-elements per iteration, two named buffer
// sets that alternate roles -> no cross-buffer register copies in steady
// state). Each compute8 waits only on loads issued one full iteration ago
// while the other buffer's 12 loads are in flight.
// ---------------------------------------------------------------------------
template<int K, int LD>
__device__ __forceinline__ void mma_loop_32x64(const __bf16* __restrict__ aPtr0,
                                               const __bf16* __restrict__ aPtr1,
                                               const __bf16* __restrict__ bPtr,
                                               v8f acc[2][4]) {
    static_assert(K % 64 == 0, "K must be a multiple of 64");
    v16bf a0[2], b0[4], a1[2], b1[4];
    load_frags<LD>(aPtr0, aPtr1, bPtr, 0,  a0, b0);
    load_frags<LD>(aPtr0, aPtr1, bPtr, 32, a1, b1);

    #pragma unroll 1
    for (int k0 = 64; k0 < K; k0 += 64) {
        compute8(acc, a0, b0);
        load_frags<LD>(aPtr0, aPtr1, bPtr, k0, a0, b0);
        __builtin_prefetch(aPtr0 + k0 + 1024, 0, 1);
        compute8(acc, a1, b1);
        load_frags<LD>(aPtr0, aPtr1, bPtr, k0 + 32, a1, b1);
        __builtin_prefetch(aPtr1 + k0 + 1024, 0, 1);
    }

    compute8(acc, a0, b0);   // drain
    compute8(acc, a1, b1);
}

// ---------------------------------------------------------------------------
// GEMM1:  q = vec @ wq_w^T + wq_b   (M=16384, N=1024, K=1024, bf16 WMMA)
// Epilogue stores q TRANSPOSED per batch: qT[b][h][s]  (bf16) so that BOTH
// operands of GEMM2 are K(=s)-contiguous.
// ---------------------------------------------------------------------------
__global__ __launch_bounds__(256)
void gemm1_wmma_kernel(const __bf16* __restrict__ A,
                       const __bf16* __restrict__ Bw,
                       const float*  __restrict__ bias,
                       __bf16* __restrict__ qT) {
    const int lane  = threadIdx.x;           // 0..31
    const int wave  = threadIdx.y;           // 0..7
    const int row   = lane & 15;
    const int hi    = lane >> 4;
    const int mTile = (blockIdx.y * 8 + wave) * 32;
    const int nTile = blockIdx.x * 64;

    const int kA = hi ? 16 : 0;              // A fragment K base
    const int kB = hi ? 16 : 0;              // B fragment K base

    const __bf16* aPtr0 = A  + (size_t)(mTile + row)      * INDIM + kA;
    const __bf16* aPtr1 = A  + (size_t)(mTile + 16 + row) * INDIM + kA;
    const __bf16* bPtr  = Bw + (size_t)(nTile + row)      * INDIM + kB;

    v8f acc[2][4] = {};
    mma_loop_32x64<INDIM, INDIM>(aPtr0, aPtr1, bPtr, acc);

    // Epilogue: add bias, convert, store transposed.
    // C layout: VGPR r <-> M = mTile + m*16 + 8*hi + r,  N = nTile + t*16 + row.
    #pragma unroll
    for (int m = 0; m < 2; ++m) {
        const int m0 = mTile + m * 16 + 8 * hi;  // 8 consecutive s values
        const int b  = m0 >> 11;                 // / SS
        const int s  = m0 & (SS - 1);
        #pragma unroll
        for (int t = 0; t < 4; ++t) {
            const int   n  = nTile + t * 16 + row;
            const float bn = bias[n];
            v8bf o;
            #pragma unroll
            for (int r = 0; r < 8; ++r) o[r] = f2bf(acc[m][t][r] + bn);
            *(v8bf*)(qT + ((size_t)b * HID + n) * SS + s) = o;   // 128-bit store
        }
    }
}

// ---------------------------------------------------------------------------
// GEMM2 + fused epilogue:
//   ret[b] = qT[b] @ qT[b]^T  (M=N=1024, K=S=2048)   [bf16 WMMA, f32 acc]
//   out[b][x] += sum_y relu(ret[x][y]) * lin_w[y]    (atomic f32 add)
// ---------------------------------------------------------------------------
__global__ __launch_bounds__(256)
void gemm2_wmma_kernel(const __bf16* __restrict__ qT,
                       const float*  __restrict__ lin_w,
                       float* __restrict__ out) {
    const int lane  = threadIdx.x;
    const int wave  = threadIdx.y;
    const int row   = lane & 15;
    const int hi    = lane >> 4;
    const int batch = blockIdx.z;
    const int mTile = (blockIdx.y * 8 + wave) * 32;
    const int nTile = blockIdx.x * 64;

    const int kA = hi ? 16 : 0;
    const int kB = hi ? 16 : 0;

    const __bf16* base  = qT + (size_t)batch * HID * SS;
    const __bf16* aPtr0 = base + (size_t)(mTile + row)      * SS + kA;
    const __bf16* aPtr1 = base + (size_t)(mTile + 16 + row) * SS + kA;
    const __bf16* bPtr  = base + (size_t)(nTile + row)      * SS + kB;

    v8f acc[2][4] = {};
    mma_loop_32x64<SS, SS>(aPtr0, aPtr1, bPtr, acc);

    // Fused: relu -> * lin_w[y] -> reduce over the 64 columns of this tile.
    float wn[4];
    #pragma unroll
    for (int t = 0; t < 4; ++t) wn[t] = lin_w[nTile + t * 16 + row];

    #pragma unroll
    for (int m = 0; m < 2; ++m) {
        const int m0 = mTile + m * 16 + 8 * hi;
        #pragma unroll
        for (int r = 0; r < 8; ++r) {
            float sum = 0.f;
            #pragma unroll
            for (int t = 0; t < 4; ++t) {
                float v = acc[m][t][r];
                sum += (v > 0.f ? v : 0.f) * wn[t];
            }
            // reduce across the 16 lanes of this half-wave (same m, different n)
            sum += __shfl_xor(sum, 1, 32);
            sum += __shfl_xor(sum, 2, 32);
            sum += __shfl_xor(sum, 4, 32);
            sum += __shfl_xor(sum, 8, 32);
            if (row == 0) {
                __hip_atomic_fetch_add(&out[batch * HID + m0 + r], sum,
                                       __ATOMIC_RELAXED, __HIP_MEMORY_SCOPE_AGENT);
            }
        }
    }
}

// ---------------------------------------------------------------------------
// Launch
// ---------------------------------------------------------------------------
extern "C" void kernel_launch(void* const* d_in, const int* in_sizes, int n_in,
                              void* d_out, int out_size, void* d_ws, size_t ws_size,
                              hipStream_t stream) {
    const float* vec   = (const float*)d_in[0];   // [8,2048,1024]
    const float* wq_w  = (const float*)d_in[1];   // [1024,1024]
    const float* wq_b  = (const float*)d_in[2];   // [1024]
    const float* lin_w = (const float*)d_in[3];   // [1,1024]
    const float* lin_b = (const float*)d_in[4];   // [1]
    float*       out   = (float*)d_out;           // [8,1024]

    // workspace layout
    char*   ws     = (char*)d_ws;
    __bf16* vec_bf = (__bf16*)ws;                                   // 32 MiB
    __bf16* wq_bf  = (__bf16*)(ws + (size_t)MTOT * INDIM * 2);      //  2 MiB
    __bf16* qT     = (__bf16*)(ws + (size_t)MTOT * INDIM * 2
                                  + (size_t)HID  * INDIM * 2);      // 32 MiB

    // 1) convert inputs to bf16
    {
        int n4 = (MTOT * INDIM) / 4;
        cvt_bf16_kernel<<<(n4 + 255) / 256, 256, 0, stream>>>(vec, vec_bf, n4);
        int w4 = (HID * INDIM) / 4;
        cvt_bf16_kernel<<<(w4 + 255) / 256, 256, 0, stream>>>(wq_w, wq_bf, w4);
    }

    // 2) init output accumulators with lin_b
    init_out_kernel<<<(BB * HID + 255) / 256, 256, 0, stream>>>(out, lin_b);

    // 3) GEMM1: q = vec @ wq^T + b, stored transposed per batch
    {
        dim3 grid(HID / 64, MTOT / 256, 1);   // (16, 64)
        dim3 block(32, 8, 1);
        gemm1_wmma_kernel<<<grid, block, 0, stream>>>(vec_bf, wq_bf, wq_b, qT);
    }

    // 4) GEMM2 + fused relu/matvec epilogue
    {
        dim3 grid(HID / 64, HID / 256, BB);   // (16, 4, 8)
        dim3 block(32, 8, 1);
        gemm2_wmma_kernel<<<grid, block, 0, stream>>>(qT, lin_w, out);
    }
}
